// CGConvLayer_47974784696372
// MI455X (gfx1250) — matrix operand order repaired and checked by reading, not compiled, not run on hardware
//
#include <hip/hip_runtime.h>

#define D 128
#define BN_EPS 0.001f

typedef __attribute__((ext_vector_type(16))) __bf16 v16bf;
typedef __attribute__((ext_vector_type(2)))  __bf16 v2bf;
typedef __attribute__((ext_vector_type(8)))  float  v8f;

__device__ __forceinline__ __bf16 to_bf16(float f) {
  // round-to-nearest-even fp32 -> bf16 (used only in tiny prep kernel)
  unsigned u = __builtin_bit_cast(unsigned, f);
  unsigned r = u + 0x7FFFu + ((u >> 16) & 1u);
  return __builtin_bit_cast(__bf16, (unsigned short)(r >> 16));
}

// Pack two fp32 -> two bf16 in one dword.
// Prefer the hw packed convert if the toolchain exposes it; otherwise
// round-half-up (+0x8000) and pack both high halves with a single V_PERM_B32.
__device__ __forceinline__ v2bf pk_bf16(float a, float b) {
#if __has_builtin(__builtin_amdgcn_cvt_pk_bf16_f32)
  return __builtin_amdgcn_cvt_pk_bf16_f32(a, b);
#else
  unsigned ua = __builtin_bit_cast(unsigned, a) + 0x8000u;
  unsigned ub = __builtin_bit_cast(unsigned, b) + 0x8000u;
  // dst bytes: [0,1] = ua bytes [2,3]; [2,3] = ub bytes [2,3]
  unsigned packed = __builtin_amdgcn_perm(ub, ua, 0x07060302u);
  return __builtin_bit_cast(v2bf, packed);
#endif
}

__device__ __forceinline__ v8f wmma_bf16(v16bf a, v16bf b, v8f c) {
  return __builtin_amdgcn_wmma_f32_16x16x32_bf16(
      /*neg_a=*/false, a, /*neg_b=*/false, b,
      /*c_mod=*/(short)0, c, /*reuse_a=*/false, /*reuse_b=*/false);
}

// A-fragment (16x32 bf16): lane = row (lane%16); half 0 holds K kb*32+0..15,
// half 1 holds K kb*32+16..31. Built from fp32 row with perm-packed converts.
__device__ __forceinline__ v16bf load_a_frag(const float* __restrict__ rp, int kb, int half) {
  const float* p0 = rp + kb * 32 + half * 16;
  const float* p1 = p0 + 8;
  float4 x0 = *reinterpret_cast<const float4*>(p0);
  float4 x1 = *reinterpret_cast<const float4*>(p0 + 4);
  float4 x2 = *reinterpret_cast<const float4*>(p1);
  float4 x3 = *reinterpret_cast<const float4*>(p1 + 4);
  union { v2bf p[8]; v16bf v; } u;
  u.p[0] = pk_bf16(x0.x, x0.y); u.p[1] = pk_bf16(x0.z, x0.w);
  u.p[2] = pk_bf16(x1.x, x1.y); u.p[3] = pk_bf16(x1.z, x1.w);
  u.p[4] = pk_bf16(x2.x, x2.y); u.p[5] = pk_bf16(x2.z, x2.w);
  u.p[6] = pk_bf16(x3.x, x3.y); u.p[7] = pk_bf16(x3.z, x3.w);
  return u.v;
}

// B-fragment (32x16 bf16): lane = col (lane%16); half 0 holds K kb*32+0..15,
// half 1 holds K kb*32+16..31. Wt is bf16 [n][k] row-major -> one 32B contiguous load.
__device__ __forceinline__ v16bf load_b_frag(const __bf16* __restrict__ Wt, int col, int kb, int half) {
  return *reinterpret_cast<const v16bf*>(Wt + col * D + kb * 32 + half * 16);
}

// ---------------- prep: transpose + convert W_node/W_gate/W_filt to bf16 [n][k] ----
__global__ void prep_weights_kernel(const float* __restrict__ Wn,
                                    const float* __restrict__ Wg,
                                    const float* __restrict__ Wf,
                                    __bf16* __restrict__ out) {
  int t = blockIdx.x * blockDim.x + threadIdx.x;
  if (t >= 3 * D * D) return;
  int which = t / (D * D);
  int r = t % (D * D);
  int n = r / D, k = r % D;
  const float* W = which == 0 ? Wn : (which == 1 ? Wg : Wf);
  out[t] = to_bf16(W[k * D + n]);
}

__global__ void zero_kernel(float* __restrict__ p, int n) {
  int i = blockIdx.x * blockDim.x + threadIdx.x;
  if (i < n) p[i] = 0.0f;
}

// ---------------- node_hidden = node_features @ W_node + b_node ---------------------
template <bool FULL>
__device__ __forceinline__ void node_gemm_body(
    const float* __restrict__ X, const __bf16* __restrict__ Wt,
    const float* __restrict__ bias, float* __restrict__ H, int nrows) {
  const int lane  = threadIdx.x & 31;
  const int wave  = threadIdx.x >> 5;
  const int half  = lane >> 4;
  const int lm    = lane & 15;
  const int mbase = blockIdx.x * 64 + (wave & 3) * 16;
  const int nbase = (wave >> 2) * 64;

  int arow = mbase + lm;
  if (!FULL && arow >= nrows) arow = nrows - 1;   // clamp loads; stores masked below
  const float* rp = X + (long)arow * D;

  v16bf afrag[4];
#pragma unroll
  for (int kb = 0; kb < 4; ++kb) afrag[kb] = load_a_frag(rp, kb, half);

#pragma unroll
  for (int nt = 0; nt < 4; ++nt) {
    const int col = nbase + nt * 16 + lm;
    v8f acc = 0.0f;
#pragma unroll
    for (int kb = 0; kb < 4; ++kb)
      acc = wmma_bf16(afrag[kb], load_b_frag(Wt, col, kb, half), acc);
    const float bv = bias[col];
#pragma unroll
    for (int r = 0; r < 8; ++r) {
      int row = mbase + r + half * 8;
      if (FULL || row < nrows) H[(long)row * D + col] = acc[r] + bv;
    }
  }
}

__global__ __launch_bounds__(256) void node_gemm_kernel(
    const float* __restrict__ X, const __bf16* __restrict__ Wt,
    const float* __restrict__ bias, float* __restrict__ H, int nrows) {
  if (blockIdx.x * 64 + 64 <= (unsigned)nrows)     // block-uniform branch
    node_gemm_body<true>(X, Wt, bias, H, nrows);
  else
    node_gemm_body<false>(X, Wt, bias, H, nrows);
}

// -------- fused edge path: gate/filt GEMMs + sigmoid*filt*gather + scatter-add ------
template <bool FULL>
__device__ __forceinline__ void edge_body(
    const float* __restrict__ E, const __bf16* __restrict__ WtG,
    const __bf16* __restrict__ WtF, const float* __restrict__ bg,
    const float* __restrict__ bfv_, const float* __restrict__ H,
    const int* __restrict__ eidx, float* __restrict__ sums,
    float* __restrict__ counts, int e0, int nedges) {
  // per-edge degree counts (one lane per edge in tile)
  if (threadIdx.x < 64) {
    int e = e0 + threadIdx.x;
    if (FULL || e < nedges) unsafeAtomicAdd(&counts[eidx[2 * e]], 1.0f);
  }

  const int lane  = threadIdx.x & 31;
  const int wave  = threadIdx.x >> 5;
  const int half  = lane >> 4;
  const int lm    = lane & 15;
  const int mloc  = (wave & 3) * 16;
  const int nbase = (wave >> 2) * 64;

  int er = e0 + mloc + lm;
  if (!FULL && er >= nedges) er = nedges - 1;
  const float* rp = E + (long)er * D;

  v16bf afrag[4];
#pragma unroll
  for (int kb = 0; kb < 4; ++kb) afrag[kb] = load_a_frag(rp, kb, half);

  int dsts[8], outs[8], valid[8];
#pragma unroll
  for (int r = 0; r < 8; ++r) {
    int e = e0 + mloc + r + half * 8;
    valid[r] = FULL || (e < nedges);
    if (!FULL && e >= nedges) e = nedges - 1;
    outs[r] = eidx[2 * e];
    dsts[r] = eidx[2 * e + 1];
  }

#pragma unroll
  for (int nt = 0; nt < 4; ++nt) {
    const int col = nbase + nt * 16 + lm;
    v8f ag = 0.0f, af = 0.0f;
#pragma unroll
    for (int kb = 0; kb < 4; ++kb) {
      ag = wmma_bf16(afrag[kb], load_b_frag(WtG, col, kb, half), ag);
      af = wmma_bf16(afrag[kb], load_b_frag(WtF, col, kb, half), af);
    }
    const float bgv = bg[col];
    const float bfv = bfv_[col];
#pragma unroll
    for (int r = 0; r < 8; ++r) {
      if (FULL || valid[r]) {
        float g  = 1.0f / (1.0f + __expf(-(ag[r] + bgv)));
        float f  = af[r] + bfv;
        float nb = H[(long)dsts[r] * D + col];           // L2-resident gather
        unsafeAtomicAdd(&sums[(long)outs[r] * D + col], g * f * nb);
      }
    }
  }
}

__global__ __launch_bounds__(256) void edge_kernel(
    const float* __restrict__ E, const __bf16* __restrict__ WtG,
    const __bf16* __restrict__ WtF, const float* __restrict__ bg,
    const float* __restrict__ bfv_, const float* __restrict__ H,
    const int* __restrict__ eidx, float* __restrict__ sums,
    float* __restrict__ counts, int nedges) {
  const int e0 = blockIdx.x * 64;
  if (e0 + 64 <= nedges)                              // block-uniform branch
    edge_body<true>(E, WtG, WtF, bg, bfv_, H, eidx, sums, counts, e0, nedges);
  else
    edge_body<false>(E, WtG, WtF, bg, bfv_, H, eidx, sums, counts, e0, nedges);
}

// ---------------- mean aggregate + residual + batchnorm + relu (in place, float4) ---
__global__ void finalize_kernel(float4* __restrict__ HO, const float4* __restrict__ sums,
                                const float* __restrict__ counts,
                                const float4* __restrict__ gamma, const float4* __restrict__ beta,
                                const float4* __restrict__ mean, const float4* __restrict__ var,
                                int n) {
  int i = blockIdx.x * blockDim.x + threadIdx.x;     // one float4 per thread
  if (i >= n * (D / 4)) return;
  int c4 = i & (D / 4 - 1);
  float  c = counts[i >> 5];
  float inv = (c > 0.0f) ? 1.0f / fmaxf(c, 1.0f) : 0.0f;
  float4 s = sums[i];
  float4 h = HO[i];
  float4 mu = mean[c4], va = var[c4], ga = gamma[c4], be = beta[c4];
  float4 o;
  o.x = fmaxf((h.x + s.x * inv - mu.x) * rsqrtf(va.x + BN_EPS) * ga.x + be.x, 0.0f);
  o.y = fmaxf((h.y + s.y * inv - mu.y) * rsqrtf(va.y + BN_EPS) * ga.y + be.y, 0.0f);
  o.z = fmaxf((h.z + s.z * inv - mu.z) * rsqrtf(va.z + BN_EPS) * ga.z + be.z, 0.0f);
  o.w = fmaxf((h.w + s.w * inv - mu.w) * rsqrtf(va.w + BN_EPS) * ga.w + be.w, 0.0f);
  HO[i] = o;
}

extern "C" void kernel_launch(void* const* d_in, const int* in_sizes, int n_in,
                              void* d_out, int out_size, void* d_ws, size_t ws_size,
                              hipStream_t stream) {
  const float* node_f = (const float*)d_in[0];
  const float* edge_f = (const float*)d_in[1];
  const float* Wn     = (const float*)d_in[2];
  const float* bnod   = (const float*)d_in[3];
  const float* Wg     = (const float*)d_in[4];
  const float* bg     = (const float*)d_in[5];
  const float* Wf     = (const float*)d_in[6];
  const float* bf     = (const float*)d_in[7];
  const float* gamma  = (const float*)d_in[8];
  const float* beta   = (const float*)d_in[9];
  const float* mean   = (const float*)d_in[10];
  const float* var    = (const float*)d_in[11];
  const int*   eidx   = (const int*)d_in[12];

  float* H = (float*)d_out;                 // holds node_hidden, then final output
  const int nnodes = in_sizes[0] / D;
  const int nedges = in_sizes[1] / D;

  // workspace layout: [3*D*D bf16 transposed weights][sums f32 nnodes*D][counts f32 nnodes]
  __bf16* wt    = (__bf16*)d_ws;
  float*  sums  = (float*)((char*)d_ws + (size_t)3 * D * D * sizeof(__bf16));
  float*  cnts  = sums + (size_t)nnodes * D;

  prep_weights_kernel<<<(3 * D * D + 255) / 256, 256, 0, stream>>>(Wn, Wg, Wf, wt);

  const int zn = nnodes * D + nnodes;
  zero_kernel<<<(zn + 255) / 256, 256, 0, stream>>>(sums, zn);

  node_gemm_kernel<<<(nnodes + 63) / 64, 256, 0, stream>>>(node_f, wt, bnod, H, nnodes);

  edge_kernel<<<(nedges + 63) / 64, 256, 0, stream>>>(
      edge_f, wt + D * D, wt + 2 * D * D, bg, bf, H, eidx, sums, cnts, nedges);

  finalize_kernel<<<(nnodes * (D / 4) + 255) / 256, 256, 0, stream>>>(
      (float4*)H, (const float4*)sums, cnts,
      (const float4*)gamma, (const float4*)beta,
      (const float4*)mean, (const float4*)var, nnodes);
}
